// TreeDecoder_66623532696068
// MI455X (gfx1250) — matrix operand order repaired and checked by reading, not compiled, not run on hardware
//
#include <hip/hip_runtime.h>

// ---------------------------------------------------------------------------
// MI455X (gfx1250) fused TreeDecoder.
// Memory-bound: ~185MB unavoidable traffic (W_up dominates) @23.3TB/s => ~8us
// floor. Fw0/Fw1 chain is linear => collapsed on device to one 128x128 matrix
// (k_combine). All GEMMs via v_wmma_f32_16x16x32_bf16 (f32 accumulate).
// This revision: native bf16 converts (v_cvt_*bf16*) instead of manual RNE
// bit-twiddling, vectorized LDS staging, Uw3^T prepacked in bf16 once.
// ---------------------------------------------------------------------------

typedef __attribute__((ext_vector_type(16))) __bf16 v16bf;
typedef __attribute__((ext_vector_type(8)))  float  v8f;

union ABFrag { v16bf v; unsigned u[8]; };
union DFrag  { v8f   v; float    f[8]; };

// native fp32 -> bf16 (single element, bits out)
__device__ __forceinline__ unsigned short bfbits(float x) {
    union { __bf16 h; unsigned short s; } r;
    r.h = (__bf16)x;
    return r.s;
}
// native fp32 x2 -> packed bf16x2 (lets clang form v_cvt_pk_bf16_f32)
__device__ __forceinline__ unsigned pk2bf(float lo, float hi) {
    union { unsigned u; __bf16 h[2]; } r;
    r.h[0] = (__bf16)lo;
    r.h[1] = (__bf16)hi;
    return r.u;
}

// A fragment (16x32 bf16) from LDS tile stored row-major [rows][kld],
// lanes row M=lane&15; VGPR v holds K = 16h + 2v, 2v+1 (matches ldsB).
__device__ __forceinline__ v16bf ldsA(const unsigned short* t, int kld,
                                      int row0, int k0, int lane) {
    int r = row0 + (lane & 15);
    int h = lane >> 4;
    const unsigned short* p = t + r * kld + k0 + 16 * h;
    ABFrag f;
#pragma unroll
    for (int v = 0; v < 8; ++v) {
        f.u[v]     = *(const unsigned*)(p + 2 * v);
    }
    return f.v;
}

// B fragment (32x16 bf16) from LDS tile holding B^T, i.e. [ncols][kld]
// (N-major). Lane = column N; VGPR v holds K = 16h + 2v, 2v+1.
__device__ __forceinline__ v16bf ldsB(const unsigned short* bt, int kld,
                                      int n0, int k0, int lane) {
    int n = n0 + (lane & 15);
    int h = lane >> 4;
    const unsigned short* p = bt + n * kld + k0 + 16 * h;
    ABFrag f;
#pragma unroll
    for (int v = 0; v < 8; ++v) f.u[v] = *(const unsigned*)(p + 2 * v);
    return f.v;
}

__device__ __forceinline__ v8f wmma_bf16(v16bf a, v16bf b, v8f c) {
    return __builtin_amdgcn_wmma_f32_16x16x32_bf16(
        /*neg_a=*/false, a, /*neg_b=*/false, b,
        /*c_mod=*/(short)0, c, /*reuse_a=*/false, /*reuse_b=*/false);
}

// ---------------------------------------------------------------------------
// Kernel 0: Fcomb = Fw0[128,1280] @ Fw1[1280,128] -> stored transposed bf16.
//           bc[j] = Fb1[j] + sum_h Fb0[h]*Fw1[h,j]
//           Also prepacks U3T = Uw3^T in bf16 (N-major) for k_node.
// ---------------------------------------------------------------------------
__global__ __launch_bounds__(256)
void k_combine(const float* __restrict__ Fw0, const float* __restrict__ Fb0,
               const float* __restrict__ Fw1, const float* __restrict__ Fb1,
               const float* __restrict__ Uw3,
               unsigned short* __restrict__ FcT, float* __restrict__ bc,
               unsigned short* __restrict__ U3T) {
    __shared__ unsigned short lA[128 * 32];   // Fw0 chunk, row-major
    __shared__ unsigned short lBt[128 * 32];  // Fw1 chunk, N-major (B^T)
    int tid = threadIdx.x, lane = tid & 31, w = tid >> 5;

    // prepack Uw3^T (bf16, N-major) into workspace; pairs over fast j axis
    for (int idx = tid; idx < 128 * 64; idx += 256) {
        int k = idx >> 6, jp = (idx & 63) * 2;
        float2 t = *(const float2*)&Uw3[k * 128 + jp];
        U3T[jp * 128 + k]       = bfbits(t.x);
        U3T[(jp + 1) * 128 + k] = bfbits(t.y);
    }

    v8f zero = {};
    v8f acc[8];
#pragma unroll
    for (int i = 0; i < 8; ++i) acc[i] = zero;

    for (int kc = 0; kc < 40; ++kc) {            // K = 1280 in chunks of 32
        __syncthreads();
        // A chunk: pairs along kk (global-fast == LDS-fast) -> packed b32 store
        for (int idx = tid; idx < 128 * 16; idx += 256) {
            int m = idx >> 4, kp = (idx & 15) * 2;
            float2 t = *(const float2*)&Fw0[m * 1280 + kc * 32 + kp];
            ((unsigned*)lA)[idx] = pk2bf(t.x, t.y);
        }
        // B^T chunk: pairs along j (global-fast), two b16 stores
        for (int idx = tid; idx < 32 * 64; idx += 256) {
            int kk = idx >> 6, jp = (idx & 63) * 2;
            float2 t = *(const float2*)&Fw1[(kc * 32 + kk) * 128 + jp];
            lBt[jp * 32 + kk]       = bfbits(t.x);
            lBt[(jp + 1) * 32 + kk] = bfbits(t.y);
        }
        __syncthreads();
        v16bf a = ldsA(lA, 32, 16 * w, 0, lane);  // wave w owns M-tile w
#pragma unroll
        for (int nt = 0; nt < 8; ++nt) {
            v16bf b = ldsB(lBt, 32, 16 * nt, 0, lane);
            acc[nt] = wmma_bf16(a, b, acc[nt]);
        }
    }

    int h = lane >> 4, nl = lane & 15;
#pragma unroll
    for (int nt = 0; nt < 8; ++nt) {
        DFrag d; d.v = acc[nt];
#pragma unroll
        for (int v = 0; v < 8; ++v) {
            int m = 16 * w + 8 * h + v;
            int j = 16 * nt + nl;
            FcT[j * 128 + m] = bfbits(d.f[v]);   // store transposed (N-major)
        }
    }
    if (tid < 128) {                             // combined bias (tiny)
        float s = Fb1[tid];
        for (int hh = 0; hh < 1280; ++hh) s += Fb0[hh] * Fw1[hh * 128 + tid];
        bc[tid] = s;
    }
}

// ---------------------------------------------------------------------------
// Kernel 1: QS[b, m, :] = sum over levels 0..2 of tree_l[b, m>>shift] @ Uw_l
//           + (Ub0+Ub1+Ub2), at level-2 granularity (m in 0..127).
// One block per batch b. M=128 gathered rows, K=3x128, N=128.
// ---------------------------------------------------------------------------
__global__ __launch_bounds__(256)
void k_ancestor(const float* __restrict__ t0, const float* __restrict__ t1,
                const float* __restrict__ t2,
                const float* __restrict__ Uw0, const float* __restrict__ Ub0,
                const float* __restrict__ Uw1, const float* __restrict__ Ub1,
                const float* __restrict__ Uw2, const float* __restrict__ Ub2,
                float* __restrict__ QS) {
    __shared__ unsigned short lA[128 * 128];  // gathered tree rows (32KB)
    __shared__ unsigned short lBt[128 * 32];  // Uw chunk, N-major (8KB)
    __shared__ float lub[128];

    int b = blockIdx.x;
    int tid = threadIdx.x, lane = tid & 31, w = tid >> 5;
    if (tid < 128) lub[tid] = Ub0[tid] + Ub1[tid] + Ub2[tid];

    v8f zero = {};
    v8f acc[8];
#pragma unroll
    for (int i = 0; i < 8; ++i) acc[i] = zero;

    for (int lvl = 0; lvl < 3; ++lvl) {
        const float* tp = (lvl == 0) ? t0 : (lvl == 1 ? t1 : t2);
        const float* up = (lvl == 0) ? Uw0 : (lvl == 1 ? Uw1 : Uw2);
        int nn    = (lvl == 0) ? 1 : (lvl == 1 ? 16 : 128);
        int shift = (lvl == 0) ? 7 : (lvl == 1 ? 3 : 0);

        __syncthreads();  // previous level's lA reads complete
        // gathered A rows: pairs along k -> packed b32 stores
        for (int idx = tid; idx < 128 * 64; idx += 256) {
            int m = idx >> 6, kp = (idx & 63) * 2;
            float2 t = *(const float2*)&tp[(b * nn + (m >> shift)) * 128 + kp];
            ((unsigned*)lA)[idx] = pk2bf(t.x, t.y);
        }
        for (int ks = 0; ks < 4; ++ks) {
            __syncthreads();
            for (int idx = tid; idx < 32 * 64; idx += 256) {
                int kk = idx >> 6, jp = (idx & 63) * 2;
                float2 t = *(const float2*)&up[(32 * ks + kk) * 128 + jp];
                lBt[jp * 32 + kk]       = bfbits(t.x);
                lBt[(jp + 1) * 32 + kk] = bfbits(t.y);
            }
            __syncthreads();
            v16bf a = ldsA(lA, 128, 16 * w, 32 * ks, lane);
#pragma unroll
            for (int nt = 0; nt < 8; ++nt) {
                v16bf bb = ldsB(lBt, 32, 16 * nt, 0, lane);
                acc[nt] = wmma_bf16(a, bb, acc[nt]);
            }
        }
    }

    int h = lane >> 4, nl = lane & 15;
#pragma unroll
    for (int nt = 0; nt < 8; ++nt) {
        DFrag d; d.v = acc[nt];
#pragma unroll
        for (int v = 0; v < 8; ++v) {
            int m = 16 * w + 8 * h + v;
            int j = 16 * nt + nl;
            QS[((size_t)b * 128 + m) * 128 + j] = d.f[v] + lub[j];
        }
    }
}

// ---------------------------------------------------------------------------
// Kernel 2: one block per leaf node n (1024 blocks).
//  phase1: anc[b,:]  = T3@Uw3 + Ub3 + QS[b, n>>3]          (LDS fp32)
//  phase2: U[2b+d,:] = T3 @ W_up[n][:, 64-col chunks]       (LDS bf16)
//  phase3: out       = LeakyReLU(U@Fcomb + bc + anc + b_bias)
// ---------------------------------------------------------------------------
__global__ __launch_bounds__(256)
void k_node(const float* __restrict__ t3,
            const unsigned short* __restrict__ U3T,   // Uw3^T bf16 (prepacked)
            const float* __restrict__ Ub3, const float* __restrict__ Wup,
            const float* __restrict__ bbias,
            const unsigned short* __restrict__ FcT,
            const float* __restrict__ bc,
            const float* __restrict__ QS, float* __restrict__ out) {
    __shared__ float          lAnc[32 * 128];   // 16KB fp32 ancestor rows
    __shared__ float          lbc[128];
    __shared__ unsigned short lT3[32 * 128];    // 8KB  tree3[:,n,:] bf16
    __shared__ unsigned short lU[64 * 128];     // 16KB upsample rows r=2b+d
    __shared__ unsigned short lDyn[64 * 128];   // 16KB reused B^T staging

    int n = blockIdx.x;
    int tid = threadIdx.x, lane = tid & 31, w = tid >> 5;
    int h = lane >> 4, nl = lane & 15;
    v8f zero = {};

    // T3 tile: pairs along k -> packed b32 stores
    for (int idx = tid; idx < 32 * 64; idx += 256) {
        int bb = idx >> 6, kp = (idx & 63) * 2;
        float2 t = *(const float2*)&t3[((size_t)bb * 1024 + n) * 128 + kp];
        ((unsigned*)lT3)[idx] = pk2bf(t.x, t.y);
    }
    if (tid < 128) lbc[tid] = bc[tid];

    // ---- phase 1: ancestor rows. wave w: M-tile w&1, N-tiles {2(w>>1), +1}
    {
        int mt = w & 1, nt0 = (w >> 1) * 2;
        v8f a0 = zero, a1 = zero;
        for (int ks = 0; ks < 4; ++ks) {
            __syncthreads();
            // U3T chunk (already bf16, N-major): straight b128 copy
            for (int idx = tid; idx < 512; idx += 256) {   // 512 x uint4 = 8KB
                int j = idx >> 2, q = idx & 3;
                ((uint4*)lDyn)[idx] =
                    *(const uint4*)(U3T + j * 128 + 32 * ks + 8 * q);
            }
            __syncthreads();
            v16bf a = ldsA(lT3, 128, 16 * mt, 32 * ks, lane);
            a0 = wmma_bf16(a, ldsB(lDyn, 32, 16 * nt0, 0, lane), a0);
            a1 = wmma_bf16(a, ldsB(lDyn, 32, 16 * (nt0 + 1), 0, lane), a1);
        }
        int anc2 = n >> 3;  // level-2 ancestor
#pragma unroll
        for (int t = 0; t < 2; ++t) {
            DFrag d; d.v = (t == 0) ? a0 : a1;
#pragma unroll
            for (int v = 0; v < 8; ++v) {
                int bb = 16 * mt + 8 * h + v;
                int j  = 16 * (nt0 + t) + nl;
                lAnc[bb * 128 + j] =
                    d.f[v] + Ub3[j] + QS[((size_t)bb * 128 + anc2) * 128 + j];
            }
        }
    }

    // ---- phase 2: upsample, 4 chunks of 64 output columns
    for (int nc = 0; nc < 4; ++nc) {
        __syncthreads();  // lDyn free (phase1 / previous chunk done)
        // W_up chunk, transposed: pairs over col (global-fast)
        for (int idx = tid; idx < 128 * 32; idx += 256) {
            int k = idx >> 5, cp = (idx & 31) * 2;
            float2 t =
                *(const float2*)&Wup[((size_t)n * 128 + k) * 256 + 64 * nc + cp];
            lDyn[cp * 128 + k]       = bfbits(t.x);
            lDyn[(cp + 1) * 128 + k] = bfbits(t.y);
        }
        if (nc < 3) {  // hint next chunk of W_up into cache
            __builtin_prefetch(
                &Wup[((size_t)n * 128 + (tid >> 6)) * 256 + 64 * (nc + 1) + (tid & 63)],
                0, 1);
        }
        __syncthreads();
        int mt = w >> 2, ntl = w & 3;  // 2 M-tiles x 4 N-tiles = 8 waves
        v8f acc = zero;
#pragma unroll
        for (int ks = 0; ks < 4; ++ks) {
            v16bf a  = ldsA(lT3, 128, 16 * mt, 32 * ks, lane);
            v16bf bb = ldsB(lDyn, 128, 16 * ntl, 32 * ks, lane);
            acc = wmma_bf16(a, bb, acc);
        }
        DFrag d; d.v = acc;
#pragma unroll
        for (int v = 0; v < 8; ++v) {
            int bb   = 16 * mt + 8 * h + v;          // batch index
            int colg = 64 * nc + 16 * ntl + nl;      // 0..255
            int dd   = colg >> 7, f = colg & 127;
            lU[(2 * bb + dd) * 128 + f] = bfbits(d.f[v]);
        }
    }

    // ---- phase 3: collapsed MLP + epilogue. wave w: M-tile w>>1, 4 N-tiles
    {
        int mt = w >> 1, nt0 = (w & 1) * 4;
        v8f acc[4];
#pragma unroll
        for (int i = 0; i < 4; ++i) acc[i] = zero;
        for (int ks = 0; ks < 4; ++ks) {
            __syncthreads();
            // FcT chunk (bf16): straight b128 copy
            for (int idx = tid; idx < 512; idx += 256) {
                int j = idx >> 2, q = idx & 3;
                ((uint4*)lDyn)[idx] =
                    *(const uint4*)(FcT + j * 128 + 32 * ks + 8 * q);
            }
            __syncthreads();
            v16bf a = ldsA(lU, 128, 16 * mt, 32 * ks, lane);
#pragma unroll
            for (int t = 0; t < 4; ++t) {
                v16bf bb = ldsB(lDyn, 32, 16 * (nt0 + t), 0, lane);
                acc[t] = wmma_bf16(a, bb, acc[t]);
            }
        }
#pragma unroll
        for (int t = 0; t < 4; ++t) {
            DFrag d; d.v = acc[t];
#pragma unroll
            for (int v = 0; v < 8; ++v) {
                int r  = 16 * mt + 8 * h + v;        // 0..63 = 2b+d
                int bb = r >> 1, dd = r & 1;
                int j  = 16 * (nt0 + t) + nl;
                float val = d.f[v] + lbc[j] + lAnc[bb * 128 + j]
                          + bbias[(size_t)(2 * n + dd) * 128 + j];
                val = (val >= 0.f) ? val : 0.2f * val;
                out[((size_t)bb * 2048 + 2 * n + dd) * 128 + j] = val;
            }
        }
    }
}

// ---------------------------------------------------------------------------
extern "C" void kernel_launch(void* const* d_in, const int* in_sizes, int n_in,
                              void* d_out, int out_size, void* d_ws, size_t ws_size,
                              hipStream_t stream) {
    (void)in_sizes; (void)n_in; (void)out_size; (void)ws_size;
    const float* t0  = (const float*)d_in[0];
    const float* t1  = (const float*)d_in[1];
    const float* t2  = (const float*)d_in[2];
    const float* t3  = (const float*)d_in[3];
    const float* Uw0 = (const float*)d_in[4];
    const float* Ub0 = (const float*)d_in[5];
    const float* Uw1 = (const float*)d_in[6];
    const float* Ub1 = (const float*)d_in[7];
    const float* Uw2 = (const float*)d_in[8];
    const float* Ub2 = (const float*)d_in[9];
    const float* Uw3 = (const float*)d_in[10];
    const float* Ub3 = (const float*)d_in[11];
    const float* Wup = (const float*)d_in[12];
    const float* Fw0 = (const float*)d_in[13];
    const float* Fb0 = (const float*)d_in[14];
    const float* Fw1 = (const float*)d_in[15];
    const float* Fb1 = (const float*)d_in[16];
    const float* bb  = (const float*)d_in[17];
    float* out = (float*)d_out;

    unsigned char* ws = (unsigned char*)d_ws;
    unsigned short* FcT = (unsigned short*)ws;            // 32 KB bf16 N-major
    float* bc  = (float*)(ws + 32768);                    // 512 B
    float* QS  = (float*)(ws + 33280);                    // 2 MB [32][128][128] f32
    unsigned short* U3T = (unsigned short*)(ws + 33280 + 2097152);  // 32 KB

    k_combine <<<1,    256, 0, stream>>>(Fw0, Fb0, Fw1, Fb1, Uw3, FcT, bc, U3T);
    k_ancestor<<<32,   256, 0, stream>>>(t0, t1, t2, Uw0, Ub0, Uw1, Ub1, Uw2, Ub2, QS);
    k_node    <<<1024, 256, 0, stream>>>(t3, U3T, Ub3, Wup, bb, FcT, bc, QS, out);
}
